// KMeansPP_12103217840140
// MI455X (gfx1250) — matrix operand order repaired and checked by reading, not compiled, not run on hardware
//
#include <hip/hip_runtime.h>
#include <climits>

// ---------------------------------------------------------------------------
// k-means++ seeding for data[N=400000, D=128] f32, k=64 on MI455X (gfx1250).
// Memory-bound (AI ~= 0.5 flop/byte): bf16-cache the data once in workspace
// (halves per-step traffic; 102.4MB fits the 192MB L2, so steady-state sweeps
// stream from L2), feed v_wmma_f32_16x16x32_bf16 with the center broadcast
// across B's 16 columns, and fuse distance-update + block-sum + weighted
// sampling per step into one kernel (last-block atomic, graph-capture safe).
// This revision: __builtin_amdgcn_sched_barrier(0) between the staged 8x
// global_load_b128 and the 4 WMMAs so the compiler emits staged
// s_wait_loadcnt (6/4/2/0) instead of load/wait/wmma lockstep.
// ---------------------------------------------------------------------------

typedef __attribute__((ext_vector_type(16))) __bf16 v16bf;
typedef __attribute__((ext_vector_type(8)))  __bf16 v8bf;
typedef __attribute__((ext_vector_type(8)))  float  v8f;

#define RPB 512   // rows (points) handled per block in the step kernel
#define TPB 256   // threads per block (8 wave32s)

#if defined(__has_builtin)
#if __has_builtin(__builtin_amdgcn_sched_barrier)
#define SCHED_BARRIER() __builtin_amdgcn_sched_barrier(0)
#endif
#endif
#ifndef SCHED_BARRIER
#define SCHED_BARRIER()
#endif

__device__ __forceinline__ unsigned short f2bf(float f) {
  unsigned u = __builtin_bit_cast(unsigned, f);
  u += 0x7fffu + ((u >> 16) & 1u);       // round-to-nearest-even
  return (unsigned short)(u >> 16);
}

__device__ __forceinline__ float rng01(unsigned s) {
  unsigned x = (s ^ 0x2Au) * 0x9E3779B9u + 0x85EBCA6Bu;
  x ^= x >> 16; x *= 0x7feb352du;
  x ^= x >> 15; x *= 0x846ca68bu;
  x ^= x >> 16;
  return (float)(x >> 8) * (1.0f / 16777216.0f);
}

__device__ __forceinline__ float waveRedAdd(float v) {  // wave32 butterfly
  v += __shfl_xor(v, 16);
  v += __shfl_xor(v, 8);
  v += __shfl_xor(v, 4);
  v += __shfl_xor(v, 2);
  v += __shfl_xor(v, 1);
  return v;
}

// --- pass 0: squared norms, closest=+inf, optional bf16 cache of data -------
template<bool CACHED>
__global__ void __launch_bounds__(TPB) kmpp_prep(
    const float* __restrict__ data, unsigned short* __restrict__ cache,
    float* __restrict__ x2, float* __restrict__ closest, int N) {
  const int lane = threadIdx.x & 31;
  const int gw = (int)((blockIdx.x * blockDim.x + threadIdx.x) >> 5);
  const int nw = (int)((gridDim.x * blockDim.x) >> 5);
  for (int row = gw; row < N; row += nw) {            // one wave per row
    const float4 f =
        *reinterpret_cast<const float4*>(data + (size_t)row * 128 + lane * 4);
    if (CACHED) {
      unsigned lo = (unsigned)f2bf(f.x) | ((unsigned)f2bf(f.y) << 16);
      unsigned hi = (unsigned)f2bf(f.z) | ((unsigned)f2bf(f.w) << 16);
      reinterpret_cast<uint2*>(cache + (size_t)row * 128)[lane] =
          make_uint2(lo, hi);
    }
    float sq = f.x * f.x + f.y * f.y + f.z * f.z + f.w * f.w;
    sq = waveRedAdd(sq);
    if (lane == 0) { x2[row] = sq; closest[row] = 3.4e38f; }
  }
}

// --- pass 1: uniform first center, zero the arrival counters ----------------
__global__ void __launch_bounds__(TPB) kmpp_seed(
    const float* __restrict__ data, const float* __restrict__ x2,
    float* __restrict__ out0, unsigned short* __restrict__ centerBf,
    float* __restrict__ c2ptr, int* __restrict__ counters, int N, int k) {
  const int t = threadIdx.x;
  if (t < k) counters[t] = 0;
  __shared__ int sh_idx;
  if (t == 0) {
    int idx = (int)(rng01(0x9999u) * (float)N);
    if (idx >= N) idx = N - 1;
    if (idx < 0) idx = 0;
    sh_idx = idx;
  }
  __syncthreads();
  const int idx = sh_idx;
  if (t < 128) {
    float val = data[(size_t)idx * 128 + t];
    out0[t] = val;
    centerBf[t] = f2bf(val);
  }
  if (t == 0) *c2ptr = x2[idx];
}

// --- per-step fused kernel: distance update + block sums + sampling ---------
template<bool CACHED>
__global__ void __launch_bounds__(TPB) kmpp_step(
    const float* __restrict__ data, const unsigned short* __restrict__ cache,
    const float* __restrict__ x2, float* __restrict__ closest,
    float* __restrict__ partials, unsigned short* __restrict__ centerBf,
    float* __restrict__ c2ptr, int* __restrict__ counters,
    float* __restrict__ outRow, int s, int N) {
  const int t = threadIdx.x;
  const int lane = t & 31;
  const int wave = t >> 5;
  const int b = (int)blockIdx.x;
  const bool hiHalf = (lane & 16) != 0;
  const int c0 = hiHalf ? 8 : 0;           // K-segment start within chunk
  const int mrow = lane & 15;              // A-matrix M = lane % 16

  __shared__ float s_red[TPB / 32];
  __shared__ int   sh_last;
  __shared__ int   sh_b2, sh_pos;
  __shared__ float sh_r;
  __shared__ float ps[TPB];

  const float c2 = *c2ptr;

  // B fragment: 32x16 bf16, every column = center chunk (broadcast).
  // Same K order as A: lane<16 holds K=kbase+0..7,16..23, lane>=16 K=kbase+8..15,24..31,
  // i.e. two 16B loads per lane per chunk.
  const __bf16* cbf = reinterpret_cast<const __bf16*>(centerBf);
  v16bf Bf[4];
#pragma unroll
  for (int tt = 0; tt < 4; ++tt) {
    v8bf blo = *reinterpret_cast<const v8bf*>(cbf + tt * 32 + c0);
    v8bf bhi = *reinterpret_cast<const v8bf*>(cbf + tt * 32 + c0 + 16);
    Bf[tt] = __builtin_shufflevector(blo, bhi, 0, 1, 2, 3, 4, 5, 6, 7, 8, 9, 10, 11, 12, 13, 14, 15);
  }

  float laneSum = 0.0f;
#pragma unroll 1
  for (int tile = wave; tile < RPB / 16; tile += TPB / 32) {
    const int r0 = b * RPB + tile * 16;
    if (r0 >= N) continue;                 // wave-uniform branch
    const int rowc = min(r0 + mrow, N - 1);

    // Prefetch this wave's next tile (128 rows ahead) into cache hierarchy.
    {
      const int rnext = min(r0 + (TPB / 32) * 16 + mrow, N - 1);
      if (CACHED)
        __builtin_prefetch(reinterpret_cast<const __bf16*>(cache) +
                               (size_t)rnext * 128 + c0, 0, 3);
      else
        __builtin_prefetch(data + (size_t)rnext * 128 + c0, 0, 3);
    }

    // Stage all four 16x32 A-fragments (8x b128 loads), then fence the
    // scheduler so the WMMAs stay below: loads return in order, so the
    // backend can use staged s_wait_loadcnt and overlap WMMA t with the
    // tail of loads t+1..3.
    v16bf Afrag[4];
#pragma unroll
    for (int tt = 0; tt < 4; ++tt) {
      if (CACHED) {
        const __bf16* prow = reinterpret_cast<const __bf16*>(cache) +
                             (size_t)rowc * 128 + tt * 32 + c0;
        v8bf lo = *reinterpret_cast<const v8bf*>(prow);        // K seg0 (16B)
        v8bf hi = *reinterpret_cast<const v8bf*>(prow + 16);   // K seg1 (16B)
#pragma unroll
        for (int i = 0; i < 8; ++i) { Afrag[tt][i] = lo[i]; Afrag[tt][i + 8] = hi[i]; }
      } else {
        const float* prow = data + (size_t)rowc * 128 + tt * 32 + c0;
#pragma unroll
        for (int i = 0; i < 8; ++i) {
          Afrag[tt][i]     = __builtin_bit_cast(__bf16, f2bf(prow[i]));
          Afrag[tt][i + 8] = __builtin_bit_cast(__bf16, f2bf(prow[16 + i]));
        }
      }
    }
    SCHED_BARRIER();
    v8f acc = {};
#pragma unroll
    for (int tt = 0; tt < 4; ++tt)         // K = 4 * 32 = 128
      acc = __builtin_amdgcn_wmma_f32_16x16x32_bf16(
          false, Afrag[tt], false, Bf[tt], (short)0, acc, false, false);

    // D layout: VGPR v holds M = v (lanes 0-15) or v+8 (lanes 16-31); all 16
    // columns are identical, so let lane (M%16 == v) own row r0+v(+8).
#pragma unroll
    for (int v = 0; v < 8; ++v) {
      const int rr = r0 + v + (hiHalf ? 8 : 0);
      if (mrow == v && rr < N) {
        float d2 = x2[rr] - 2.0f * acc[v] + c2;
        d2 = fmaxf(d2, 0.0f);
        float nv = fminf(closest[rr], d2);
        closest[rr] = nv;
        laneSum += nv;
      }
    }
  }

  // block reduction of updated-closest sum -> partials[b]
  float wsum = waveRedAdd(laneSum);
  if (lane == 0) s_red[wave] = wsum;
  __syncthreads();
  if (t == 0) {
    float bs = 0.0f;
#pragma unroll
    for (int i = 0; i < TPB / 32; ++i) bs += s_red[i];
    partials[b] = bs;
    __threadfence();
    int arrived = atomicAdd(&counters[s], 1);
    sh_last = (arrived == (int)gridDim.x - 1) ? 1 : 0;
  }
  __syncthreads();
  if (!sh_last) return;
  __threadfence();                         // acquire all blocks' writes

  // ---- sampling phase: only the last-arriving block runs this ----
  const int numBlocks = (int)gridDim.x;
  if (t == 0) {
    float total = 0.0f;
    for (int j = 0; j < numBlocks; ++j) total += partials[j];
    float u = rng01((unsigned)s) * total;
    float run = 0.0f, r = u;
    int b2 = numBlocks - 1, found = 0;
    for (int j = 0; j < numBlocks; ++j) {
      float pj = partials[j];
      if (!found && pj > 0.0f && run + pj >= u) { b2 = j; r = u - run; found = 1; }
      run += pj;
    }
    if (!found) {
      for (int j = numBlocks - 1; j >= 0; --j)
        if (partials[j] > 0.0f) { b2 = j; break; }
      r = 3.4e38f;                         // forces last-element fallback
    }
    sh_b2 = b2; sh_r = r; sh_pos = INT_MAX;
  }
  __syncthreads();
  const int rbase = sh_b2 * RPB;
  const float r = sh_r;
  const int i0 = rbase + 2 * t, i1 = i0 + 1;
  const float e0 = (i0 < N) ? closest[i0] : 0.0f;
  const float e1 = (i1 < N) ? closest[i1] : 0.0f;
  const float pairv = e0 + e1;
  ps[t] = pairv;
  __syncthreads();
  for (int off = 1; off < TPB; off <<= 1) {   // Hillis-Steele inclusive scan
    float add = (t >= off) ? ps[t - off] : 0.0f;
    __syncthreads();
    ps[t] += add;
    __syncthreads();
  }
  const float excl = ps[t] - pairv;
  int pos = INT_MAX;
  if (i0 < N && excl + e0 >= r)            pos = 2 * t;
  else if (i1 < N && excl + e0 + e1 >= r)  pos = 2 * t + 1;
  if (pos != INT_MAX) atomicMin(&sh_pos, pos);
  __syncthreads();
  const int idx = (sh_pos == INT_MAX) ? min(N - 1, rbase + RPB - 1)
                                      : (rbase + sh_pos);
  if (t < 128) {                           // emit centroid + next center
    float val = data[(size_t)idx * 128 + t];
    outRow[t] = val;
    centerBf[t] = f2bf(val);
  }
  if (t == 0) *c2ptr = x2[idx];
}

// ---------------------------------------------------------------------------
extern "C" void kernel_launch(void* const* d_in, const int* in_sizes, int n_in,
                              void* d_out, int out_size, void* d_ws,
                              size_t ws_size, hipStream_t stream) {
  (void)n_in;
  const int D = 128;
  const float* data = (const float*)d_in[0];
  const int N = in_sizes[0] / D;
  const int k = out_size / D;
  float* out = (float*)d_out;

  // workspace carve-out (256B aligned slices)
  char* base = (char*)d_ws;
  size_t off = 0;
  auto carve = [&](size_t bytes) -> void* {
    void* p = base + off;
    off += (bytes + 255) & ~(size_t)255;
    return p;
  };
  float* closest            = (float*)carve((size_t)N * 4);
  float* x2                 = (float*)carve((size_t)N * 4);
  const int numBlocks       = (N + RPB - 1) / RPB;
  float* partials           = (float*)carve((size_t)numBlocks * 4);
  unsigned short* centerBf  = (unsigned short*)carve((size_t)D * 2);
  float* c2ptr              = (float*)carve(4);
  int* counters             = (int*)carve((size_t)k * 4);
  unsigned short* cache     = (unsigned short*)(base + off);
  const bool cached = (off + (size_t)N * D * 2) <= ws_size;

  if (cached)
    kmpp_prep<true><<<1024, TPB, 0, stream>>>(data, cache, x2, closest, N);
  else
    kmpp_prep<false><<<1024, TPB, 0, stream>>>(data, cache, x2, closest, N);

  kmpp_seed<<<1, TPB, 0, stream>>>(data, x2, out, centerBf, c2ptr, counters,
                                   N, k);

  for (int s = 1; s < k; ++s) {
    if (cached)
      kmpp_step<true><<<numBlocks, TPB, 0, stream>>>(
          data, cache, x2, closest, partials, centerBf, c2ptr, counters,
          out + (size_t)s * D, s, N);
    else
      kmpp_step<false><<<numBlocks, TPB, 0, stream>>>(
          data, cache, x2, closest, partials, centerBf, c2ptr, counters,
          out + (size_t)s * D, s, N);
  }
}
